// TrajectoryAttention_51230369906750
// MI455X (gfx1250) — matrix operand order repaired and checked by reading, not compiled, not run on hardware
//
#include <hip/hip_runtime.h>
#include <hip/hip_bf16.h>

// ---------------------------------------------------------------------------
// Trajectory attention for MI455X (gfx1250, wave32, WMMA).
// All GEMMs in bf16 via v_wmma_f32_16x16x32_bf16, f32 accumulation.
// GEMMs stage A/B K-slabs into LDS with GLOBAL_LOAD_ASYNC_TO_LDS_B128
// (ASYNCcnt-tracked, double buffered), WMMA operands come from ds_load_b128.
// ---------------------------------------------------------------------------

typedef __bf16 bf16_t;
typedef __attribute__((ext_vector_type(16))) __bf16 v16bf;
typedef __attribute__((ext_vector_type(8)))  float  v8f;
typedef __attribute__((ext_vector_type(4)))  int    v4i;

// problem constants (compile-time)
constexpr int Bb = 4;     // batch
constexpr int Ff = 8;     // frames
constexpr int Pp = 196;   // tokens per frame
constexpr int Nn = Ff * Pp;   // 1568
constexpr int Cc = 768;   // channels
constexpr int Hh = 12;    // heads
constexpr int Dd = 64;    // head dim
constexpr int BH = Bb * Hh;   // 48
constexpr int PPAD = 224; // padded keys per frame (multiple of 32)

union ABFrag { v16bf v; uint4 q[2]; };
union CFrag  { v8f  v; float f[8]; };

__device__ inline v8f zero8() {
  v8f z = {0.f,0.f,0.f,0.f,0.f,0.f,0.f,0.f};
  return z;
}

// A-fragment (16x32 bf16, MxK): per-lane 16 contiguous K elems.
// caller passes p = &A[row(lane)][k0 + hiA], hiA = lane&16
__device__ inline v16bf ld_gap16(const bf16_t* p) {
  ABFrag f;
  f.q[0] = *reinterpret_cast<const uint4*>(p);
  f.q[1] = *reinterpret_cast<const uint4*>(p + 8);
  return f.v;
}
// B-fragment (32x16 bf16, KxN): per-lane 16 contiguous K elems.
// caller passes p = &Bt[col(lane)][k0 + (lane&16)]
__device__ inline v16bf ld_cont16(const bf16_t* p) {
  ABFrag f;
  f.q[0] = *reinterpret_cast<const uint4*>(p);
  f.q[1] = *reinterpret_cast<const uint4*>(p + 8);
  return f.v;
}

__device__ inline v8f wmma_bf16(v16bf a, v16bf b, v8f c) {
  return __builtin_amdgcn_wmma_f32_16x16x32_bf16(
      /*neg_a=*/false, a, /*neg_b=*/false, b,
      /*c_mod=*/(short)0, c, /*reuse_a=*/false, /*reuse_b=*/false);
}

// ---------------------------------------------------------------------------
// CDNA5 async copy global->LDS (16B per lane), ASYNCcnt tracked.
// Builtin signature (probe-learned): (v4i AS1*, v4i AS3*, imm offset, imm cpol)
// ---------------------------------------------------------------------------
__device__ inline void async_copy_b128(bf16_t* lds_row, int byte_off,
                                       const bf16_t* gsrc) {
  char* g = const_cast<char*>(reinterpret_cast<const char*>(gsrc)) + byte_off;
  char* l = reinterpret_cast<char*>(lds_row) + byte_off;
#if __has_builtin(__builtin_amdgcn_global_load_async_to_lds_b128)
  __builtin_amdgcn_global_load_async_to_lds_b128(
      (__attribute__((address_space(1))) v4i*)g,
      (__attribute__((address_space(3))) v4i*)l, 0, 0);
#else
  unsigned loff = (unsigned)(__UINTPTR_TYPE__)(__attribute__((address_space(3))) char*)l;
  asm volatile("global_load_async_to_lds_b128 %0, %1, off"
               :: "v"(loff), "v"(g) : "memory");
#endif
}

__device__ inline void wait_async0() {
#if __has_builtin(__builtin_amdgcn_s_wait_asynccnt)
  __builtin_amdgcn_s_wait_asynccnt(0);
#else
  asm volatile("s_wait_asynccnt 0x0" ::: "memory");
#endif
}

// ---------------------------------------------------------------------------
// prep: f32 -> bf16 conversions
// ---------------------------------------------------------------------------
__global__ void __launch_bounds__(256)
convert_f32_bf16(const float* __restrict__ in, bf16_t* __restrict__ out, int n) {
  int i = blockIdx.x * 256 + threadIdx.x;
  if (i < n) out[i] = (bf16_t)in[i];
}

// W[K][Nout] (f32) -> Wt[Nout][K] (bf16): WMMA B operand wants K contiguous.
__global__ void __launch_bounds__(256)
transpose_to_bf16(const float* __restrict__ W, bf16_t* __restrict__ Wt,
                  int K, int Nout) {
  int i = blockIdx.x * 256 + threadIdx.x;
  if (i < K * Nout) {
    int k = i / Nout, n = i - k * Nout;
    Wt[(size_t)n * K + k] = (bf16_t)W[i];
  }
}

// ---------------------------------------------------------------------------
// generic bf16 GEMM: C[M][NOUT] = A[M][768] * Bt[NOUT][768]^T + bias
// AMAP:  0 = identity rows, 1 = diagonal-trajectory gather from o[B,N,F,C]
// OMODE: 0 = qkv -> split q/k[BH][N][d] + vT[BH][d][F][PPAD]
//        1 = q2  -> [BH][N][d], pre-scaled by 1/8
//        2 = kv2 -> bf16 row-major [M][1536]
//        3 = final -> f32 row-major [M][768]
// block = 8 waves, block tile 64x128, wave tile 32x32 (2x2 fragments),
// K staged through double-buffered LDS via async copies.
// ---------------------------------------------------------------------------
template<int AMAP>
__device__ inline const bf16_t* a_row_ptr(const bf16_t* A, int row) {
  if (AMAP == 0) {
    return A + (size_t)row * Cc;
  } else {
    int b  = row / Nn, fp = row - b * Nn;
    int f  = fp / Pp;
    return A + ((size_t)(b * Nn + fp) * Ff + f) * Cc;  // o[b, fp, f, :]
  }
}

template<int OMODE>
__device__ inline void store_cfrag(v8f cv, int rbase, int cbase, int lane,
                                   const float* __restrict__ bias,
                                   float* __restrict__ outf,
                                   bf16_t* __restrict__ outb,
                                   bf16_t* __restrict__ qo,
                                   bf16_t* __restrict__ ko,
                                   bf16_t* __restrict__ vto) {
  CFrag c; c.v = cv;
  int col = cbase + (lane & 15);
  int m0  = rbase + ((lane & 16) >> 1);
  float bv = bias[col];
#pragma unroll
  for (int r = 0; r < 8; ++r) {
    int row = m0 + r;
    float val = c.f[r] + bv;
    if (OMODE == 3) {
      outf[(size_t)row * Cc + col] = val;
    } else if (OMODE == 2) {
      outb[(size_t)row * (2 * Cc) + col] = (bf16_t)val;
    } else if (OMODE == 1) {
      int b = row / Nn, nq = row - b * Nn;
      int hh = col >> 6, dd = col & 63;
      outb[((size_t)(b * Hh + hh) * Nn + nq) * Dd + dd] = (bf16_t)(val * 0.125f);
    } else {  // OMODE == 0, qkv split
      int b = row / Nn, nq = row - b * Nn;
      int sel = col / Cc, cc = col - sel * Cc;
      int hh = cc >> 6, dd = cc & 63;
      size_t idx = ((size_t)(b * Hh + hh) * Nn + nq) * Dd + dd;
      if (sel == 0) {
        qo[idx] = (bf16_t)val;
      } else if (sel == 1) {
        ko[idx] = (bf16_t)val;
      } else {
        int fr = nq / Pp, pp = nq - fr * Pp;
        vto[(((size_t)(b * Hh + hh) * Dd + dd) * Ff + fr) * PPAD + pp] = (bf16_t)val;
      }
    }
  }
}

template<int AMAP, int OMODE>
__global__ void __launch_bounds__(256)
gemm_bf16_kernel(const bf16_t* __restrict__ A, const bf16_t* __restrict__ Bt,
                 const float* __restrict__ bias,
                 float* __restrict__ outf, bf16_t* __restrict__ outb,
                 bf16_t* __restrict__ qo, bf16_t* __restrict__ ko,
                 bf16_t* __restrict__ vto) {
  constexpr int LDK = 40;  // 32 + 8 pad elems (80 B rows, 16B-aligned)
  __shared__ bf16_t As[2][64][LDK];
  __shared__ bf16_t Bs[2][128][LDK];

  const int t    = threadIdx.x;
  const int lane = t & 31;
  const int wv   = t >> 5;
  const int wm   = wv >> 2, wn = wv & 3;
  const int row0 = blockIdx.y * 64;
  const int col0 = blockIdx.x * 128;
  const int hiA  = lane & 16;         // +16 elements for upper half-lanes
  const int hiB  = lane & 16;         // +16 elements for upper half-lanes

  // copy roles: thread t moves one 16B chunk of the A slab and two of B
  const int crow = t >> 2;             // 0..63
  const int coff = (t & 3) * 16;       // byte chunk within a 64B K-slab row
  const bf16_t* ag  = a_row_ptr<AMAP>(A, row0 + crow);
  const bf16_t* bg0 = Bt + (size_t)(col0 + crow) * Cc;
  const bf16_t* bg1 = Bt + (size_t)(col0 + crow + 64) * Cc;

  v8f c00 = zero8(), c01 = zero8(), c10 = zero8(), c11 = zero8();

  // prologue: stage K-step 0 into buffer 0
  async_copy_b128(&As[0][crow][0],      coff, ag);
  async_copy_b128(&Bs[0][crow][0],      coff, bg0);
  async_copy_b128(&Bs[0][crow + 64][0], coff, bg1);

  for (int step = 0; step < Cc / 32; ++step) {
    const int buf = step & 1;
    wait_async0();       // this wave's copies into As/Bs[buf] are in LDS
    __syncthreads();     // everyone's copies visible; prior reads of buf^1 done

    if (step + 1 < Cc / 32) {  // stage next K-step into the other buffer
      const int kk = (step + 1) * 32;
      async_copy_b128(&As[buf ^ 1][crow][0],      coff, ag  + kk);
      async_copy_b128(&Bs[buf ^ 1][crow][0],      coff, bg0 + kk);
      async_copy_b128(&Bs[buf ^ 1][crow + 64][0], coff, bg1 + kk);
    }

    v16bf a0  = ld_gap16(&As[buf][wm * 32 +      (lane & 15)][0] + hiA);
    v16bf a1  = ld_gap16(&As[buf][wm * 32 + 16 + (lane & 15)][0] + hiA);
    v16bf bb0 = ld_cont16(&Bs[buf][wn * 32 +      (lane & 15)][0] + hiB);
    v16bf bb1 = ld_cont16(&Bs[buf][wn * 32 + 16 + (lane & 15)][0] + hiB);

    c00 = wmma_bf16(a0, bb0, c00);
    c01 = wmma_bf16(a0, bb1, c01);
    c10 = wmma_bf16(a1, bb0, c10);
    c11 = wmma_bf16(a1, bb1, c11);

    __syncthreads();     // finish reading buf before it is refilled next step
  }

  const int wr0 = row0 + wm * 32;
  const int wc0 = col0 + wn * 32;
  store_cfrag<OMODE>(c00, wr0,      wc0,      lane, bias, outf, outb, qo, ko, vto);
  store_cfrag<OMODE>(c01, wr0,      wc0 + 16, lane, bias, outf, outb, qo, ko, vto);
  store_cfrag<OMODE>(c10, wr0 + 16, wc0,      lane, bias, outf, outb, qo, ko, vto);
  store_cfrag<OMODE>(c11, wr0 + 16, wc0 + 16, lane, bias, outf, outb, qo, ko, vto);
}

// ---------------------------------------------------------------------------
// stage-1 spatial attention: one wave per (bh, 16-query tile, frame).
//   S = (q . k^T) * 1/8  over 196 keys of frame f, softmax per row,
//   O = P @ V -> o[b, nq, f, hh*64+dd] (bf16)
// ---------------------------------------------------------------------------
__global__ void __launch_bounds__(32)
spatial_attn_kernel(const bf16_t* __restrict__ qb, const bf16_t* __restrict__ kb,
                    const bf16_t* __restrict__ vT, bf16_t* __restrict__ obuf) {
  __shared__ float  sc[16][PPAD];
  __shared__ bf16_t pr[16][PPAD];

  const int bid = blockIdx.x;
  const int f   = bid & 7;
  const int qt  = (bid >> 3) % (Nn / 16);
  const int bh  = bid / (8 * (Nn / 16));
  const int b   = bh / Hh, hh = bh - b * Hh;

  const int lane = threadIdx.x;
  const int hiA  = lane & 16;

  const bf16_t* qrow   = qb + ((size_t)bh * Nn + qt * 16 + (lane & 15)) * Dd + hiA;
  const bf16_t* kfbase = kb + ((size_t)bh * Nn + f * Pp) * Dd;

  // ---- scores: 13 key tiles of 16 (196 -> 208, tail masked) ----
#pragma unroll
  for (int kt = 0; kt < 13; ++kt) {
    int key  = kt * 16 + (lane & 15);
    int keyc = key > (Pp - 1) ? (Pp - 1) : key;
    const bf16_t* krow = kfbase + (size_t)keyc * Dd + (lane & 16);
    v8f c = zero8();
#pragma unroll
    for (int dd0 = 0; dd0 < Dd; dd0 += 32) {
      v16bf a  = ld_gap16(qrow + dd0);
      v16bf bf = ld_cont16(krow + dd0);
      c = wmma_bf16(a, bf, c);
    }
    CFrag cf; cf.v = c;
    int m0 = (lane & 16) >> 1;
#pragma unroll
    for (int r = 0; r < 8; ++r)
      sc[m0 + r][key] = (key < Pp) ? cf.f[r] * 0.125f : -1e30f;
  }
  __syncthreads();

  // ---- softmax over 196 keys: lane l and l^16 split row (l&15) ----
  {
    int row = lane & 15, half = lane >> 4;
    float mx = -1e30f;
    for (int k = half; k < Pp; k += 2) mx = fmaxf(mx, sc[row][k]);
    mx = fmaxf(mx, __shfl_xor(mx, 16));
    float sum = 0.f;
    for (int k = half; k < Pp; k += 2) sum += __expf(sc[row][k] - mx);
    sum += __shfl_xor(sum, 16);
    float inv = 1.f / sum;
    for (int k = half; k < Pp; k += 2)
      pr[row][k] = (bf16_t)(__expf(sc[row][k] - mx) * inv);
    for (int k = Pp + half; k < PPAD; k += 2)
      pr[row][k] = (bf16_t)0.0f;  // zero pad -> padded keys contribute nothing
  }
  __syncthreads();

  // ---- O = P @ V : A = probs (LDS), B = vT (padded, contiguous) ----
  const bf16_t* prow = &pr[lane & 15][0] + hiA;
#pragma unroll
  for (int nt = 0; nt < 4; ++nt) {
    int dd = nt * 16 + (lane & 15);
    const bf16_t* vrow = vT + (((size_t)bh * Dd + dd) * Ff + f) * PPAD + (lane & 16);
    v8f c = zero8();
#pragma unroll
    for (int kk = 0; kk < PPAD; kk += 32) {
      v16bf a  = ld_gap16(prow + kk);
      v16bf bf = ld_cont16(vrow + kk);
      c = wmma_bf16(a, bf, c);
    }
    CFrag cf; cf.v = c;
    int m0 = (lane & 16) >> 1;
#pragma unroll
    for (int r = 0; r < 8; ++r) {
      int nq = qt * 16 + m0 + r;
      obuf[((size_t)(b * Nn + nq) * Ff + f) * Cc + hh * Dd + dd] = (bf16_t)cf.f[r];
    }
  }
}

// ---------------------------------------------------------------------------
// stage-2 temporal attention: F=8, d=64 per (b,h,n); one wave each.
// q2 is pre-scaled by 1/8.  k2v2 row = [(b*N+n)*F+f][1536]: k2 cols 0..767,
// v2 cols 768..1535.
// ---------------------------------------------------------------------------
__global__ void __launch_bounds__(256)
temporal_attn_kernel(const bf16_t* __restrict__ q2,
                     const bf16_t* __restrict__ k2v2,
                     bf16_t* __restrict__ attn) {
  const int lane = threadIdx.x & 31;
  const int t    = blockIdx.x * 8 + (threadIdx.x >> 5);
  const int b    = t / (Hh * Nn);
  const int r0   = t - b * Hh * Nn;
  const int hh   = r0 / Nn;
  const int nq   = r0 - hh * Nn;

  const bf16_t* qrow = q2 + ((size_t)(b * Hh + hh) * Nn + nq) * Dd;
  float q0 = (float)qrow[lane], q1 = (float)qrow[lane + 32];

  float sc[Ff];
#pragma unroll
  for (int f = 0; f < Ff; ++f) {
    const bf16_t* krow = k2v2 + ((size_t)(b * Nn + nq) * Ff + f) * (2 * Cc) + hh * Dd;
    float s = q0 * (float)krow[lane] + q1 * (float)krow[lane + 32];
#pragma unroll
    for (int off = 16; off > 0; off >>= 1) s += __shfl_xor(s, off);
    sc[f] = s;
  }
  float mx = sc[0];
#pragma unroll
  for (int f = 1; f < Ff; ++f) mx = fmaxf(mx, sc[f]);
  float sum = 0.f;
#pragma unroll
  for (int f = 0; f < Ff; ++f) { sc[f] = __expf(sc[f] - mx); sum += sc[f]; }
  float inv = 1.f / sum;

  float a0 = 0.f, a1 = 0.f;
#pragma unroll
  for (int f = 0; f < Ff; ++f) {
    const bf16_t* vrow =
        k2v2 + ((size_t)(b * Nn + nq) * Ff + f) * (2 * Cc) + Cc + hh * Dd;
    float w = sc[f] * inv;
    a0 += w * (float)vrow[lane];
    a1 += w * (float)vrow[lane + 32];
  }
  bf16_t* orow = attn + (size_t)(b * Nn + nq) * Cc + hh * Dd;
  orow[lane]      = (bf16_t)a0;
  orow[lane + 32] = (bf16_t)a1;
}

// ---------------------------------------------------------------------------
// launch
// ---------------------------------------------------------------------------
extern "C" void kernel_launch(void* const* d_in, const int* in_sizes, int n_in,
                              void* d_out, int out_size, void* d_ws, size_t ws_size,
                              hipStream_t stream) {
  const float* x    = (const float*)d_in[0];
  const float* Wqkv = (const float*)d_in[1];
  const float* bqkv = (const float*)d_in[2];
  const float* Wq   = (const float*)d_in[3];
  const float* bq   = (const float*)d_in[4];
  const float* Wkv  = (const float*)d_in[5];
  const float* bkv  = (const float*)d_in[6];
  const float* Wo   = (const float*)d_in[7];
  const float* bo   = (const float*)d_in[8];
  float* out = (float*)d_out;

  char* ws = (char*)d_ws;
  size_t off = 0;
  auto take = [&](size_t elems) -> bf16_t* {
    bf16_t* p = (bf16_t*)(ws + off);
    off = (off + elems * sizeof(bf16_t) + 255) & ~(size_t)255;
    return p;
  };
  bf16_t* xb     = take((size_t)Bb * Nn * Cc);            // x in bf16
  bf16_t* wqkvT  = take((size_t)3 * Cc * Cc);             // Wqkv^T
  bf16_t* wqT    = take((size_t)Cc * Cc);                 // Wq^T
  bf16_t* wkvT   = take((size_t)2 * Cc * Cc);             // Wkv^T
  bf16_t* woT    = take((size_t)Cc * Cc);                 // Wo^T
  bf16_t* qbuf   = take((size_t)BH * Nn * Dd);            // q  [BH][N][d]
  bf16_t* kbuf   = take((size_t)BH * Nn * Dd);            // k  [BH][N][d]
  bf16_t* vTbuf  = take((size_t)BH * Dd * Ff * PPAD);     // v^T[BH][d][F][224]
  bf16_t* obuf   = take((size_t)Bb * Nn * Ff * Cc);       // o  [B][N][F][C]
  bf16_t* q2buf  = take((size_t)BH * Nn * Dd);            // q2 [BH][N][d] (scaled)
  bf16_t* k2v2   = take((size_t)Bb * Nn * Ff * 2 * Cc);   // kv2 rows
  bf16_t* attnb  = take((size_t)Bb * Nn * Cc);            // stage-2 output

  // --- prep: bf16 conversions & weight transposes ---
  {
    int nx = Bb * Nn * Cc;
    convert_f32_bf16<<<(nx + 255) / 256, 256, 0, stream>>>(x, xb, nx);
    transpose_to_bf16<<<(Cc * 3 * Cc + 255) / 256, 256, 0, stream>>>(Wqkv, wqkvT, Cc, 3 * Cc);
    transpose_to_bf16<<<(Cc * Cc + 255) / 256, 256, 0, stream>>>(Wq, wqT, Cc, Cc);
    transpose_to_bf16<<<(Cc * 2 * Cc + 255) / 256, 256, 0, stream>>>(Wkv, wkvT, Cc, 2 * Cc);
    transpose_to_bf16<<<(Cc * Cc + 255) / 256, 256, 0, stream>>>(Wo, woT, Cc, Cc);
    (void)hipMemsetAsync(vTbuf, 0, (size_t)BH * Dd * Ff * PPAD * sizeof(bf16_t), stream);
  }

  // --- qkv projection: [6272,768] x [768,2304], split heads on store ---
  gemm_bf16_kernel<0, 0><<<dim3(3 * Cc / 128, Bb * Nn / 64), 256, 0, stream>>>(
      xb, wqkvT, bqkv, nullptr, nullptr, qbuf, kbuf, vTbuf);

  // --- stage 1: spatial attention, wave per (bh, qtile, frame) ---
  spatial_attn_kernel<<<BH * (Nn / 16) * Ff, 32, 0, stream>>>(qbuf, kbuf, vTbuf, obuf);

  // --- q2 = diag(o) @ Wq (gather fused via AMAP=1), pre-scaled ---
  gemm_bf16_kernel<1, 1><<<dim3(Cc / 128, Bb * Nn / 64), 256, 0, stream>>>(
      obuf, wqT, bq, nullptr, q2buf, nullptr, nullptr, nullptr);

  // --- kv2 = o @ Wkv : [50176,768] x [768,1536] (dominant GEMM) ---
  gemm_bf16_kernel<0, 2><<<dim3(2 * Cc / 128, Bb * Nn * Ff / 64), 256, 0, stream>>>(
      obuf, wkvT, bkv, nullptr, k2v2, nullptr, nullptr, nullptr);

  // --- stage 2: temporal attention over F=8 per trajectory ---
  temporal_attn_kernel<<<Bb * Hh * Nn / 8, 256, 0, stream>>>(q2buf, k2v2, attnb);

  // --- output projection -> f32 d_out ---
  gemm_bf16_kernel<0, 3><<<dim3(Cc / 128, Bb * Nn / 64), 256, 0, stream>>>(
      attnb, woT, bo, out, nullptr, nullptr, nullptr, nullptr);
}
